// UnbatchedSingleHeadTransformer_50302656971267
// MI455X (gfx1250) — matrix operand order repaired and checked by reading, not compiled, not run on hardware
//
#include <hip/hip_runtime.h>
#include <cstdint>
#include <cstddef>

// ---------------------------------------------------------------------------
// CDNA5 (gfx1250) transformer block.
//   * all GEMMs: v_wmma_f32_16x16x32_bf16, 128x128 block tile, 64x64 per wave
//   * staging: global_load_async_to_lds_b128 (ASYNCcnt) + double-buffered LDS
//   * all B operands pre-transposed ([N x K] row-major) so staging is pure
//     contiguous b128 copies; V is produced transposed by the V-proj epilogue
// ---------------------------------------------------------------------------

typedef __bf16 bf16_t;
typedef __attribute__((ext_vector_type(16))) __bf16 v16bf;
typedef __attribute__((ext_vector_type(8)))  float  v8f;

#define EP_BIAS    1
#define EP_RELU    2
#define EP_SCAUSAL 4
#define EP_RESID   8

#define LDS_STRIDE 40   // 32 + 8 pad bf16 (80B rows, conflict-free b128)
#define BT 128          // block tile (M and N)
#define KT 32           // k step

// ---- async 16B global->LDS copy (GLOBAL_LOAD_ASYNC_TO_LDS_B128, GV mode) ---
__device__ __forceinline__ void async_cp16(uint32_t lds_off, const bf16_t* g) {
  asm volatile("global_load_async_to_lds_b128 %0, %1, off"
               :: "v"(lds_off), "v"((uint64_t)(uintptr_t)g)
               : "memory");
}
__device__ __forceinline__ void wait_async_le8() {
  asm volatile("s_wait_asynccnt 0x8" ::: "memory");
}
__device__ __forceinline__ void wait_async_0() {
  asm volatile("s_wait_asynccnt 0x0" ::: "memory");
}

// A fragment (16x32 bf16, ISA 7.12.2): two contiguous 8-element runs.
__device__ __forceinline__ v16bf frag_a(const bf16_t* As, int row, int kh) {
  const bf16_t* p = As + row * LDS_STRIDE + kh * 8;
  union { uint4 u[2]; v16bf v; } c;
  c.u[0] = *reinterpret_cast<const uint4*>(p);
  c.u[1] = *reinterpret_cast<const uint4*>(p + 16);
  return c.v;
}
// B fragment (32x16 bf16) staged n-major: two contiguous 8-element runs.
__device__ __forceinline__ v16bf frag_b(const bf16_t* Bs, int col, int kh) {
  const bf16_t* p = Bs + col * LDS_STRIDE + kh * 8;
  union { uint4 u[2]; v16bf v; } c;
  c.u[0] = *reinterpret_cast<const uint4*>(p);
  c.u[1] = *reinterpret_cast<const uint4*>(p + 16);
  return c.v;
}

// C[M x N] = A[M x K] @ B[N x K]^T + epilogue.  A,B bf16 row-major.
// 128 threads = 4 waves (2x2), 64x64 per wave = 4x4 WMMA accumulators.
// M,N multiples of 128; K multiple of 32 (true for all uses here).
template <int EPI, bool OUT_BF16, bool OUT_TRANS>
__global__ __launch_bounds__(128) void gemm_wmma(
    const bf16_t* __restrict__ A, const bf16_t* __restrict__ B,
    void* __restrict__ C, const float* __restrict__ bias,
    const float* __restrict__ resid,
    int K, int lda, int ldb, int ldc, float scale) {
  __shared__ bf16_t AsBuf[2][BT * LDS_STRIDE];
  __shared__ bf16_t BsBuf[2][BT * LDS_STRIDE];

  const int tid = threadIdx.x;
  const int m0  = blockIdx.y * BT;
  const int n0  = blockIdx.x * BT;

  // Fully-masked causal block: emit -inf, skip all compute.
  if ((EPI & EP_SCAUSAL) && (n0 > m0 + BT - 1)) {
    float* Cf = reinterpret_cast<float*>(C);
    const float ninf = -__builtin_inff();
    for (int i = tid; i < BT * BT; i += 128) {
      int r = i >> 7, c = i & 127;
      Cf[(size_t)(m0 + r) * ldc + (n0 + c)] = ninf;
    }
    return;
  }

  const int lane = tid & 31;
  const int wave = tid >> 5;
  const int wm   = (wave >> 1) * 64;
  const int wn   = (wave & 1) * 64;
  const int kh   = lane >> 4;
  const int l16  = lane & 15;

  // per-thread staging assignment: 4 chunks of A + 4 of B (16B each)
  // chunk c = tid + 128*i : row = c>>2 (0..127), ch = c&3 (16B within 64B row)
  uint32_t asOff[2], bsOff[2];
  asOff[0] = (uint32_t)(uintptr_t)&AsBuf[0][0];
  asOff[1] = (uint32_t)(uintptr_t)&AsBuf[1][0];
  bsOff[0] = (uint32_t)(uintptr_t)&BsBuf[0][0];
  bsOff[1] = (uint32_t)(uintptr_t)&BsBuf[1][0];

  auto stage = [&](int k0, int buf) {
#pragma unroll
    for (int i = 0; i < 4; ++i) {
      const int c  = tid + 128 * i;
      const int r  = c >> 2;
      const int ch = c & 3;
      const uint32_t loff = (uint32_t)(r * (LDS_STRIDE * 2) + ch * 16);
      async_cp16(asOff[buf] + loff, A + (size_t)(m0 + r) * lda + k0 + ch * 8);
      async_cp16(bsOff[buf] + loff, B + (size_t)(n0 + r) * ldb + k0 + ch * 8);
    }
  };

  v8f acc[4][4];
#pragma unroll
  for (int t = 0; t < 4; ++t)
#pragma unroll
    for (int u = 0; u < 4; ++u)
#pragma unroll
      for (int i = 0; i < 8; ++i) acc[t][u][i] = 0.0f;

  const int nt = K / KT;
  stage(0, 0);

  for (int it = 0; it < nt; ++it) {
    const int cur = it & 1;
    if (it + 1 < nt) {
      stage((it + 1) * KT, cur ^ 1);
      wait_async_le8();   // oldest 8 (tile `it`) retired; next tile in flight
    } else {
      wait_async_0();
    }
    __syncthreads();      // all waves' tile `it` now visible in LDS

    const bf16_t* As = &AsBuf[cur][0];
    const bf16_t* Bs = &BsBuf[cur][0];

    v16bf bfr[4];
#pragma unroll
    for (int u = 0; u < 4; ++u) bfr[u] = frag_b(Bs, wn + u * 16 + l16, kh);
#pragma unroll
    for (int t = 0; t < 4; ++t) {
      const v16bf af = frag_a(As, wm + t * 16 + l16, kh);
#pragma unroll
      for (int u = 0; u < 4; ++u)
        acc[t][u] = __builtin_amdgcn_wmma_f32_16x16x32_bf16(
            false, af, false, bfr[u], (short)0, acc[t][u], false, false);
    }
    __syncthreads();      // protect buffer `cur` before tile it+2 overwrites
  }

  // ---- epilogue: VGPR i -> M = kh*8 + i (within 16-tile), lanes -> N ----
#pragma unroll
  for (int t = 0; t < 4; ++t) {
#pragma unroll
    for (int u = 0; u < 4; ++u) {
      const int gn = n0 + wn + u * 16 + l16;
      const int gmb = m0 + wm + t * 16 + kh * 8;
      if (OUT_TRANS) {
        // write C^T: 8 consecutive bf16 per lane -> single 16B store
        union { uint4 q; bf16_t h[8]; } pk;
#pragma unroll
        for (int i = 0; i < 8; ++i) {
          float v = acc[t][u][i];
          if (EPI & EP_BIAS) v += bias[gn];
          pk.h[i] = (bf16_t)v;
        }
        *reinterpret_cast<uint4*>(
            reinterpret_cast<bf16_t*>(C) + (size_t)gn * ldc + gmb) = pk.q;
      } else {
#pragma unroll
        for (int i = 0; i < 8; ++i) {
          const int gm = gmb + i;
          float v = acc[t][u][i];
          if (EPI & EP_BIAS) v += bias[gn];
          if (EPI & EP_SCAUSAL) {
            v *= scale;
            if (gn > gm) v = -__builtin_inff();
          }
          if (EPI & EP_RESID) v += resid[(size_t)gm * ldc + gn];
          if (EPI & EP_RELU) v = v > 0.0f ? v : 0.0f;
          if (OUT_BF16)
            reinterpret_cast<bf16_t*>(C)[(size_t)gm * ldc + gn] = (bf16_t)v;
          else
            reinterpret_cast<float*>(C)[(size_t)gm * ldc + gn] = v;
        }
      }
    }
  }
}

__global__ void cvt_f32_bf16(const float* __restrict__ in,
                             bf16_t* __restrict__ out, size_t n) {
  size_t i = (size_t)blockIdx.x * blockDim.x + threadIdx.x;
  size_t stride = (size_t)gridDim.x * blockDim.x;
  for (; i < n; i += stride) out[i] = (bf16_t)in[i];
}

// out[c][r] = (bf16) in[r][c]   (R x C fp32 -> C x R bf16), 32x32 LDS tiles
__global__ __launch_bounds__(256) void cvt_transpose_f32_bf16(
    const float* __restrict__ in, bf16_t* __restrict__ out, int R, int C) {
  __shared__ float t[32][33];
  const int bc = blockIdx.x * 32, br = blockIdx.y * 32;
  const int tx = threadIdx.x, ty = threadIdx.y;  // (32, 8)
#pragma unroll
  for (int i = 0; i < 32; i += 8)
    t[ty + i][tx] = in[(size_t)(br + ty + i) * C + bc + tx];
  __syncthreads();
#pragma unroll
  for (int i = 0; i < 32; i += 8)
    out[(size_t)(bc + ty + i) * R + br + tx] = (bf16_t)t[tx][ty + i];
}

__global__ __launch_bounds__(256) void softmax_row_kernel(
    const float* __restrict__ S, bf16_t* __restrict__ P, int N) {
  __shared__ float red[256];
  const int row = blockIdx.x;
  const float* s = S + (size_t)row * N;
  bf16_t* p = P + (size_t)row * N;

  float m = -__builtin_inff();
  for (int c = threadIdx.x; c < N; c += 256) m = fmaxf(m, s[c]);
  red[threadIdx.x] = m;
  __syncthreads();
  for (int st = 128; st > 0; st >>= 1) {
    if (threadIdx.x < st)
      red[threadIdx.x] = fmaxf(red[threadIdx.x], red[threadIdx.x + st]);
    __syncthreads();
  }
  m = red[0];
  __syncthreads();

  float sum = 0.0f;
  for (int c = threadIdx.x; c < N; c += 256) sum += __expf(s[c] - m);
  red[threadIdx.x] = sum;
  __syncthreads();
  for (int st = 128; st > 0; st >>= 1) {
    if (threadIdx.x < st) red[threadIdx.x] += red[threadIdx.x + st];
    __syncthreads();
  }
  const float inv = 1.0f / red[0];
  for (int c = threadIdx.x; c < N; c += 256)
    p[c] = (bf16_t)(__expf(s[c] - m) * inv);
}

__global__ __launch_bounds__(256) void layernorm_kernel(
    const float* __restrict__ X, const float* __restrict__ gamma,
    const float* __restrict__ beta, bf16_t* __restrict__ Y, int D) {
  __shared__ float red[256];
  const int row = blockIdx.x;
  const float* x = X + (size_t)row * D;
  bf16_t* y = Y + (size_t)row * D;

  float s = 0.0f;
  for (int c = threadIdx.x; c < D; c += 256) s += x[c];
  red[threadIdx.x] = s;
  __syncthreads();
  for (int st = 128; st > 0; st >>= 1) {
    if (threadIdx.x < st) red[threadIdx.x] += red[threadIdx.x + st];
    __syncthreads();
  }
  const float mu = red[0] / (float)D;
  __syncthreads();

  float v = 0.0f;
  for (int c = threadIdx.x; c < D; c += 256) {
    float d = x[c] - mu;
    v += d * d;
  }
  red[threadIdx.x] = v;
  __syncthreads();
  for (int st = 128; st > 0; st >>= 1) {
    if (threadIdx.x < st) red[threadIdx.x] += red[threadIdx.x + st];
    __syncthreads();
  }
  const float rstd = rsqrtf(red[0] / (float)D + 1e-5f);
  for (int c = threadIdx.x; c < D; c += 256)
    y[c] = (bf16_t)((x[c] - mu) * rstd * gamma[c] + beta[c]);
}

extern "C" void kernel_launch(void* const* d_in, const int* in_sizes, int n_in,
                              void* d_out, int out_size, void* d_ws,
                              size_t ws_size, hipStream_t stream) {
  (void)in_sizes; (void)n_in; (void)out_size; (void)ws_size;
  constexpr int N = 4096, D = 1024, F = 4096;

  const float* E     = (const float*)d_in[0];
  const float* Wq    = (const float*)d_in[1];
  const float* bq    = (const float*)d_in[2];
  const float* Wk    = (const float*)d_in[3];
  const float* bk    = (const float*)d_in[4];
  const float* Wv    = (const float*)d_in[5];
  const float* bv    = (const float*)d_in[6];
  const float* gamma = (const float*)d_in[7];
  const float* beta  = (const float*)d_in[8];
  const float* W1    = (const float*)d_in[9];
  const float* b1    = (const float*)d_in[10];
  const float* W2    = (const float*)d_in[11];
  const float* b2    = (const float*)d_in[12];
  float* out = (float*)d_out;

  // ---- workspace carve-up (256B aligned) ----
  char* ws = (char*)d_ws;
  size_t off = 0;
  auto take = [&](size_t bytes) -> char* {
    char* p = ws + off;
    off = (off + bytes + 255) & ~(size_t)255;
    return p;
  };
  bf16_t* Ebf  = (bf16_t*)take((size_t)N * D * 2);
  bf16_t* Wqt  = (bf16_t*)take((size_t)D * D * 2);  // [D][D] transposed
  bf16_t* Wkt  = (bf16_t*)take((size_t)D * D * 2);
  bf16_t* Wvt  = (bf16_t*)take((size_t)D * D * 2);
  bf16_t* W1t  = (bf16_t*)take((size_t)D * F * 2);  // [F][D]
  bf16_t* W2t  = (bf16_t*)take((size_t)F * D * 2);  // [D][F]
  bf16_t* Qbf  = (bf16_t*)take((size_t)N * D * 2);  // [N][D]
  bf16_t* Kbf  = (bf16_t*)take((size_t)N * D * 2);  // [N][D]
  bf16_t* Vt   = (bf16_t*)take((size_t)N * D * 2);  // [D][N] (transposed)
  float*  Obuf = (float*) take((size_t)N * D * 4);
  bf16_t* Xbf  = (bf16_t*)take((size_t)N * D * 2);
  float*  Sbuf = (float*) take((size_t)N * N * 4);  // scores (64 MiB)
  bf16_t* Pbf  = (bf16_t*)take((size_t)N * N * 2);  // probs  (32 MiB)
  bf16_t* Hbf  = (bf16_t*)Sbuf;  // alias: S dead after softmax

  // ---- converts (weights transposed so every GEMM uses B=[N x K]) ----
  cvt_f32_bf16<<<2048, 256, 0, stream>>>(E, Ebf, (size_t)N * D);
  {
    dim3 b(32, 8);
    cvt_transpose_f32_bf16<<<dim3(D / 32, D / 32), b, 0, stream>>>(Wq, Wqt, D, D);
    cvt_transpose_f32_bf16<<<dim3(D / 32, D / 32), b, 0, stream>>>(Wk, Wkt, D, D);
    cvt_transpose_f32_bf16<<<dim3(D / 32, D / 32), b, 0, stream>>>(Wv, Wvt, D, D);
    cvt_transpose_f32_bf16<<<dim3(F / 32, D / 32), b, 0, stream>>>(W1, W1t, D, F);
    cvt_transpose_f32_bf16<<<dim3(D / 32, F / 32), b, 0, stream>>>(W2, W2t, F, D);
  }

  const dim3 blk(128);
  // Q = E@Wq + bq, K = E@Wk + bk : [N][D] bf16
  gemm_wmma<EP_BIAS, true, false><<<dim3(D / BT, N / BT), blk, 0, stream>>>(
      Ebf, Wqt, Qbf, bq, nullptr, D, D, D, D, 0.0f);
  gemm_wmma<EP_BIAS, true, false><<<dim3(D / BT, N / BT), blk, 0, stream>>>(
      Ebf, Wkt, Kbf, bk, nullptr, D, D, D, D, 0.0f);
  // V^T = (E@Wv + bv)^T : [D][N] bf16 (vectorized transposed epilogue)
  gemm_wmma<EP_BIAS, true, true><<<dim3(D / BT, N / BT), blk, 0, stream>>>(
      Ebf, Wvt, Vt, bv, nullptr, D, D, D, N, 0.0f);

  // S = (Q @ K^T) / 32, causal (upper blocks early-exit to -inf)
  gemm_wmma<EP_SCAUSAL, false, false><<<dim3(N / BT, N / BT), blk, 0, stream>>>(
      Qbf, Kbf, Sbuf, nullptr, nullptr, D, D, D, N, 1.0f / 32.0f);

  softmax_row_kernel<<<N, 256, 0, stream>>>(Sbuf, Pbf, N);

  // O = P @ (V^T)^T + E (residual) : [N][D] f32
  gemm_wmma<EP_RESID, false, false><<<dim3(D / BT, N / BT), blk, 0, stream>>>(
      Pbf, Vt, Obuf, nullptr, E, N, N, N, D, 0.0f);

  layernorm_kernel<<<N, 256, 0, stream>>>(Obuf, gamma, beta, Xbf, D);

  // H = relu(X @ W1 + b1) : [N][F] bf16
  gemm_wmma<EP_BIAS | EP_RELU, true, false>
      <<<dim3(F / BT, N / BT), blk, 0, stream>>>(
          Xbf, W1t, Hbf, b1, nullptr, D, D, D, F, 0.0f);

  // out = H @ W2 + b2 : [N][D] f32
  gemm_wmma<EP_BIAS, false, false><<<dim3(D / BT, N / BT), blk, 0, stream>>>(
      Hbf, W2t, out, b2, nullptr, F, F, F, D, 0.0f);
}
